// IterationBlock_11854109737227
// MI455X (gfx1250) — compile-verified, matrix operand order checked by reading, not run on hardware
//
#include <hip/hip_runtime.h>
#include <hip/hip_bf16.h>

typedef __attribute__((ext_vector_type(16))) __bf16 v16bf;
typedef __attribute__((ext_vector_type(8)))  __bf16 bf16x8;
typedef __attribute__((ext_vector_type(8)))  float  v8f;

#define PI_F 3.14159265358979323846f
#define IMG 256
#define NANG 50
#define BATCH 4

// ---------------------------------------------------------------------------
// K0: ramp-filter impulse response (pre-scaled by pi/(2A)) + angle tables.
// hs[n] = (pi/100) * (1/512) * [ (1/128)*sum_{k=1}^{255} k cos(pi k n/256) + (-1)^n ]
// ---------------------------------------------------------------------------
__global__ void precompute_kernel(const float* __restrict__ angles,
                                  float* __restrict__ hs,
                                  float* __restrict__ cosT,
                                  float* __restrict__ sinT) {
  int n = threadIdx.x;  // 512 threads
  if (n < 512) {
    float s = 0.f;
    for (int k = 1; k < 256; ++k)
      s += (float)k * cosf(PI_F * (float)(k * n) * (1.0f / 256.0f));
    float parity = (n & 1) ? -1.f : 1.f;
    float r = (s * (1.0f / 128.0f) + parity) * (1.0f / 512.0f);
    hs[n] = r * (PI_F / (2.0f * (float)NANG));
  }
  if (n < NANG) {
    float a = angles[n];
    cosT[n] = cosf(a);
    sinT[n] = sinf(a);
  }
}

// ---------------------------------------------------------------------------
// K1: radon forward projection (circle-masked bilinear) minus x_sinogram.
// ---------------------------------------------------------------------------
__device__ __forceinline__ float sample_masked(const float* __restrict__ im,
                                               int y, int x) {
  if ((unsigned)y > 255u || (unsigned)x > 255u) return 0.f;
  float dy = (float)y - 127.5f, dx = (float)x - 127.5f;
  if (dy * dy + dx * dx > 128.0f * 128.0f) return 0.f;
  return im[(y << 8) + x];
}

__global__ void radon_fwd_diff_kernel(const float* __restrict__ img,
                                      const float* __restrict__ x_sino,
                                      const float* __restrict__ cosT,
                                      const float* __restrict__ sinT,
                                      float* __restrict__ diff) {
  const int a = blockIdx.x, b = blockIdx.y, s = threadIdx.x;
  const float ca = cosT[a], sa = sinT[a];
  const float c = 127.5f;
  const float sv = (float)s - c;
  const float* im = img + (b << 16);
  float acc = 0.f;
  for (int t = 0; t < IMG; ++t) {
    float tv = (float)t - c;
    float xf = sv * ca - tv * sa + c;   // column
    float yf = sv * sa + tv * ca + c;   // row
    int x0 = (int)floorf(xf), y0 = (int)floorf(yf);
    float wx = xf - (float)x0, wy = yf - (float)y0;
    acc += sample_masked(im, y0,     x0    ) * (1.f - wy) * (1.f - wx)
         + sample_masked(im, y0,     x0 + 1) * (1.f - wy) * wx
         + sample_masked(im, y0 + 1, x0    ) * wy * (1.f - wx)
         + sample_masked(im, y0 + 1, x0 + 1) * wy * wx;
  }
  int idx = (b * NANG + a) * IMG + s;
  diff[idx] = acc - x_sino[idx];
}

// ---------------------------------------------------------------------------
// K2: ramp filter as 512-point circular convolution, row + impulse in LDS.
// ---------------------------------------------------------------------------
__global__ void filter_kernel(const float* __restrict__ diff,
                              const float* __restrict__ hs,
                              float* __restrict__ filt) {
  __shared__ float row[IMG];
  __shared__ float hh[512];
  const int a = blockIdx.x, b = blockIdx.y, t = threadIdx.x;
  const int base = (b * NANG + a) * IMG;
  row[t] = diff[base + t];
  hh[t] = hs[t];
  hh[t + 256] = hs[t + 256];
  __syncthreads();
  float acc = 0.f;
  for (int s = 0; s < IMG; ++s)
    acc += row[s] * hh[(t - s + 512) & 511];
  filt[base + t] = acc;
}

// ---------------------------------------------------------------------------
// K3: backprojection + y_update, writes output #2 and NHWC concat (b,y,x,2).
// ---------------------------------------------------------------------------
__global__ void backproj_kernel(const float* __restrict__ filt,
                                const float* __restrict__ y_prev,
                                const float* __restrict__ y_cat_prev,
                                const float* __restrict__ cosT,
                                const float* __restrict__ sinT,
                                const float* __restrict__ step,
                                float* __restrict__ out_upd,
                                float* __restrict__ cat) {
  __shared__ float sf[NANG * IMG];
  const int yq = blockIdx.x, b = blockIdx.y, x = threadIdx.x;
  for (int i = threadIdx.x; i < NANG * IMG; i += blockDim.x)
    sf[i] = filt[b * NANG * IMG + i];
  __syncthreads();
  const float c = 127.5f;
  const float fx = (float)x - c, fy = (float)yq - c;
  float acc = 0.f;
  for (int a = 0; a < NANG; ++a) {
    float det = fx * cosT[a] + fy * sinT[a] + c;
    int i0 = (int)floorf(det);
    i0 = i0 < 0 ? 0 : (i0 > IMG - 2 ? IMG - 2 : i0);
    float w = det - (float)i0;
    float v = sf[a * IMG + i0] * (1.f - w) + sf[a * IMG + i0 + 1] * w;
    acc += (det >= 0.f && det <= (float)(IMG - 1)) ? v : 0.f;
  }
  float mask = (fx * fx + fy * fy <= 128.0f * 128.0f) ? 1.f : 0.f;
  int idx = (b << 16) + (yq << 8) + x;
  float upd = y_prev[idx] + step[0] * acc * mask;
  out_upd[idx] = upd;
  cat[idx * 2 + 0] = upd;
  cat[idx * 2 + 1] = y_cat_prev[idx];
}

// ---------------------------------------------------------------------------
// Pack B (weight) fragments in exact WMMA lane/element order, hi/lo split.
// Layout: [kt][nt][lane][16 elems] of bf16, per array (H and L).
// Element e of lane (n=lane_lo, half): k = 16*(e>>3) + 8*half + (e&7).
//   IC==2 : k -> ic=k&1, tap=k>>1, valid k<18 (single zero-padded K-tile)
//   IC==32: k-in-tile -> ic, tap = kt (9 K-tiles)
// ---------------------------------------------------------------------------
template <int IC, int OC>
__global__ void pack_bfrag_kernel(const float* __restrict__ w,
                                  __bf16* __restrict__ outH,
                                  __bf16* __restrict__ outL) {
  constexpr int NT = (OC + 15) / 16;
  constexpr int KT = (IC == 2) ? 1 : 9;
  const int tid = blockIdx.x * blockDim.x + threadIdx.x;
  if (tid >= KT * NT * 32) return;
  const int lane = tid & 31;
  const int nt   = (tid >> 5) % NT;
  const int kt   = tid / (32 * NT);
  const int lane_lo = lane & 15;
  const int half    = lane >> 4;
  const int n = nt * 16 + lane_lo;
#pragma unroll
  for (int e = 0; e < 16; ++e) {
    const int kk = ((e >> 3) << 4) + half * 8 + (e & 7);
    const int k  = kt * 32 + kk;
    int ic, tap;
    bool kvalid;
    if (IC == 2) { ic = k & 1; tap = k >> 1; kvalid = (k < 18); }
    else         { ic = kk;    tap = kt;     kvalid = true; }
    float v = 0.f;
    if (kvalid && n < OC) v = w[(n * IC + ic) * 9 + tap];
    const __bf16 h = (__bf16)v;
    outH[tid * 16 + e] = h;
    outL[tid * 16 + e] = (__bf16)(v - (float)h);
  }
}

__device__ __forceinline__ v8f wmma_bf16(v16bf a, v16bf b, v8f c) {
  return __builtin_amdgcn_wmma_f32_16x16x32_bf16(false, a, false, b, (short)0,
                                                 c, false, false);
}

// ---------------------------------------------------------------------------
// conv1: IC=2 (NHWC f32 cat) -> OC=32, relu, writes split-bf16 NHWC hi/lo.
// Branchless clamped loads + select (no exec-divergent loads).
// ---------------------------------------------------------------------------
__global__ __launch_bounds__(128) void conv1_wmma_kernel(
    const float* __restrict__ cat, const __bf16* __restrict__ wH,
    const __bf16* __restrict__ wL, const float* __restrict__ bias,
    __bf16* __restrict__ outH, __bf16* __restrict__ outL) {
  const int lane    = threadIdx.x & 31;
  const int wave    = threadIdx.x >> 5;
  const int lane_lo = lane & 15;
  const int half    = lane >> 4;
  const int mtile   = blockIdx.x * 4 + wave;
  const int p = mtile * 16 + lane_lo;
  const int b = p >> 16, y = (p >> 8) & 255, x = p & 255;

  v16bf ah, al;
#pragma unroll
  for (int e = 0; e < 16; ++e) {
    const int k = ((e >> 3) << 4) + half * 8 + (e & 7);  // single K-tile
    const int ic = k & 1, tap = k >> 1;
    const int ky = tap / 3, kx = tap - ky * 3;
    const int iy = y + ky - 1, ix = x + kx - 1;
    const bool valid = (k < 18) & ((unsigned)iy < 256u) & ((unsigned)ix < 256u);
    const int iyc = iy < 0 ? 0 : (iy > 255 ? 255 : iy);
    const int ixc = ix < 0 ? 0 : (ix > 255 ? 255 : ix);
    float v = cat[((((b << 16) + (iyc << 8) + ixc)) << 1) + ic];
    v = valid ? v : 0.f;
    const __bf16 h = (__bf16)v;
    ah[e] = h;
    al[e] = (__bf16)(v - (float)h);
  }

  v8f acc[2] = {};
#pragma unroll
  for (int nt = 0; nt < 2; ++nt) {
    const int fbase = (nt * 32 + lane) * 16;
    v16bf bh = *(const v16bf*)(wH + fbase);
    v16bf bl = *(const v16bf*)(wL + fbase);
    acc[nt] = wmma_bf16(ah, bl, acc[nt]);
    acc[nt] = wmma_bf16(al, bh, acc[nt]);
    acc[nt] = wmma_bf16(ah, bh, acc[nt]);
  }

#pragma unroll
  for (int nt = 0; nt < 2; ++nt) {
    const int oc = nt * 16 + lane_lo;
    const float bv = bias[oc];
#pragma unroll
    for (int j = 0; j < 8; ++j) {
      const int pm = mtile * 16 + j + 8 * half;
      float v = fmaxf(acc[nt][j] + bv, 0.f);
      const __bf16 h = (__bf16)v;
      outH[pm * 32 + oc] = h;
      outL[pm * 32 + oc] = (__bf16)(v - (float)h);
    }
  }
}

// ---------------------------------------------------------------------------
// conv2/conv3: IC=32 from pre-split bf16 NHWC, K = 9 taps x 32 ch.
// A fragment per tap: 4 aligned 16B vector loads (hi0,hi1,lo0,lo1) + select.
// NT=2 relu split-out (conv2) or NT=1 f32 NCHW out (conv3).
// ---------------------------------------------------------------------------
template <int NT, bool RELU, bool F32OUT>
__global__ __launch_bounds__(128) void conv32_wmma_kernel(
    const __bf16* __restrict__ inH, const __bf16* __restrict__ inL,
    const __bf16* __restrict__ wH,  const __bf16* __restrict__ wL,
    const float* __restrict__ bias,
    __bf16* __restrict__ outH, __bf16* __restrict__ outL,
    float* __restrict__ outF) {
  const int lane    = threadIdx.x & 31;
  const int wave    = threadIdx.x >> 5;
  const int lane_lo = lane & 15;
  const int half    = lane >> 4;
  const int mtile   = blockIdx.x * 4 + wave;
  const int p = mtile * 16 + lane_lo;
  const int b = p >> 16, y = (p >> 8) & 255, x = p & 255;

  v8f acc[NT] = {};
  const bf16x8 zero8 = {};

  for (int kt = 0; kt < 9; ++kt) {
    const int ky = kt / 3, kx = kt - ky * 3;
    const int iy = y + ky - 1, ix = x + kx - 1;
    const bool valid = ((unsigned)iy < 256u) & ((unsigned)ix < 256u);
    const int iyc = iy < 0 ? 0 : (iy > 255 ? 255 : iy);
    const int ixc = ix < 0 ? 0 : (ix > 255 ? 255 : ix);
    const int pix = (b << 16) + (iyc << 8) + ixc;
    const __bf16* baseH = inH + pix * 32 + half * 8;
    const __bf16* baseL = inL + pix * 32 + half * 8;
    bf16x8 ah0 = *(const bf16x8*)(baseH);
    bf16x8 ah1 = *(const bf16x8*)(baseH + 16);
    bf16x8 al0 = *(const bf16x8*)(baseL);
    bf16x8 al1 = *(const bf16x8*)(baseL + 16);
    ah0 = valid ? ah0 : zero8;
    ah1 = valid ? ah1 : zero8;
    al0 = valid ? al0 : zero8;
    al1 = valid ? al1 : zero8;
    v16bf ah = __builtin_shufflevector(ah0, ah1, 0, 1, 2, 3, 4, 5, 6, 7, 8, 9,
                                       10, 11, 12, 13, 14, 15);
    v16bf al = __builtin_shufflevector(al0, al1, 0, 1, 2, 3, 4, 5, 6, 7, 8, 9,
                                       10, 11, 12, 13, 14, 15);
#pragma unroll
    for (int nt = 0; nt < NT; ++nt) {
      const int fbase = ((kt * NT + nt) * 32 + lane) * 16;
      v16bf bh = *(const v16bf*)(wH + fbase);
      v16bf bl = *(const v16bf*)(wL + fbase);
      acc[nt] = wmma_bf16(ah, bl, acc[nt]);
      acc[nt] = wmma_bf16(al, bh, acc[nt]);
      acc[nt] = wmma_bf16(ah, bh, acc[nt]);
    }
  }

  if (F32OUT) {
    if (lane_lo == 0) {  // only output channel 0 exists
      const float bv = bias[0];
#pragma unroll
      for (int j = 0; j < 8; ++j) {
        const int pm = mtile * 16 + j + 8 * half;
        float v = acc[0][j] + bv;
        if (RELU) v = fmaxf(v, 0.f);
        outF[pm] = v;
      }
    }
  } else {
#pragma unroll
    for (int nt = 0; nt < NT; ++nt) {
      const int oc = nt * 16 + lane_lo;
      const float bv = bias[oc];
#pragma unroll
      for (int j = 0; j < 8; ++j) {
        const int pm = mtile * 16 + j + 8 * half;
        float v = acc[nt][j] + bv;
        if (RELU) v = fmaxf(v, 0.f);
        const __bf16 h = (__bf16)v;
        outH[pm * 32 + oc] = h;
        outL[pm * 32 + oc] = (__bf16)(v - (float)h);
      }
    }
  }
}

// ---------------------------------------------------------------------------
// Host-side launch
// ---------------------------------------------------------------------------
extern "C" void kernel_launch(void* const* d_in, const int* in_sizes, int n_in,
                              void* d_out, int out_size, void* d_ws,
                              size_t ws_size, hipStream_t stream) {
  (void)in_sizes; (void)n_in; (void)out_size; (void)ws_size;

  const float* x_sino     = (const float*)d_in[0];   // (4,1,50,256)
  const float* y_prev     = (const float*)d_in[1];   // (4,1,256,256)
  const float* y_cat_prev = (const float*)d_in[2];   // (4,1,256,256)
  const float* angles     = (const float*)d_in[3];   // (50,)
  const float* step       = (const float*)d_in[4];   // scalar
  const float* w1         = (const float*)d_in[5];   // (32,2,3,3)
  const float* b1         = (const float*)d_in[6];   // (32,)
  const float* w2         = (const float*)d_in[7];   // (32,32,3,3)
  const float* b2         = (const float*)d_in[8];   // (32,)
  const float* w3         = (const float*)d_in[9];   // (1,32,3,3)
  const float* b3         = (const float*)d_in[10];  // (1,)

  float* out_img = (float*)d_out;                // y_img (NCHW, C=1)
  float* out_upd = out_img + BATCH * IMG * IMG;  // y_img_update

  // ---- workspace carve-up ----
  const int NPIX = BATCH * IMG * IMG;  // 262144
  float* ws    = (float*)d_ws;
  float* hs    = ws;                          // 512
  float* cosT  = ws + 512;                    // 64
  float* sinT  = ws + 576;                    // 64
  float* diffb = ws + 640;                    // 51200
  float* filt  = diffb + BATCH * NANG * IMG;  // 51200
  float* cat   = filt + BATCH * NANG * IMG;   // NHWC (b,y,x,2) = 524288
  // bf16 region (starts 32B-aligned: 627328 floats = 2509312 bytes)
  __bf16* bw = (__bf16*)(cat + NPIX * 2);
  __bf16* h1H = bw;                 // NHWC (b,y,x,32) each 8388608 elems
  __bf16* h1L = h1H + NPIX * 32;
  __bf16* h2H = h1L + NPIX * 32;
  __bf16* h2L = h2H + NPIX * 32;
  __bf16* pk1H = h2L + NPIX * 32;   // 1*2*32*16 = 1024
  __bf16* pk1L = pk1H + 1024;
  __bf16* pk2H = pk1L + 1024;       // 9*2*32*16 = 9216
  __bf16* pk2L = pk2H + 9216;
  __bf16* pk3H = pk2L + 9216;       // 9*1*32*16 = 4608
  __bf16* pk3L = pk3H + 4608;

  // setup (tiny, once per call)
  precompute_kernel<<<1, 512, 0, stream>>>(angles, hs, cosT, sinT);
  pack_bfrag_kernel<2, 32><<<1, 64, 0, stream>>>(w1, pk1H, pk1L);
  pack_bfrag_kernel<32, 32><<<3, 256, 0, stream>>>(w2, pk2H, pk2L);
  pack_bfrag_kernel<32, 1><<<2, 256, 0, stream>>>(w3, pk3H, pk3L);

  // physics pipeline
  radon_fwd_diff_kernel<<<dim3(NANG, BATCH), IMG, 0, stream>>>(
      y_prev, x_sino, cosT, sinT, diffb);
  filter_kernel<<<dim3(NANG, BATCH), IMG, 0, stream>>>(diffb, hs, filt);
  backproj_kernel<<<dim3(IMG, BATCH), IMG, 0, stream>>>(
      filt, y_prev, y_cat_prev, cosT, sinT, step, out_upd, cat);

  // CNN: implicit-GEMM WMMA convs
  const int blocks = NPIX / 16 / 4;  // 4096 (4 waves/block, 16 px/wave)
  conv1_wmma_kernel<<<blocks, 128, 0, stream>>>(cat, pk1H, pk1L, b1, h1H, h1L);
  conv32_wmma_kernel<2, true, false><<<blocks, 128, 0, stream>>>(
      h1H, h1L, pk2H, pk2L, b2, h2H, h2L, nullptr);
  conv32_wmma_kernel<1, false, true><<<blocks, 128, 0, stream>>>(
      h2H, h2L, pk3H, pk3L, b3, nullptr, nullptr, out_img);
}
